// SelfAttention_44684839748299
// MI455X (gfx1250) — compile-verified, matrix operand order checked by reading, not run on hardware
//
#include <hip/hip_runtime.h>
#include <hip/hip_bf16.h>
#include <stdint.h>

typedef __attribute__((ext_vector_type(16))) _Float16 v16h;
typedef __attribute__((ext_vector_type(8)))  _Float16 v8h;
typedef __attribute__((ext_vector_type(8)))  float    v8f;

#define TOTTOK 32768   // B*N = 8*4096
#define NTOK   4096    // tokens per batch
#define CIN    256
#define CKP    32      // C/8
#define LDSPITCH 40    // padded LDS row pitch (halves): 80B -> conflict-free b128

__device__ __forceinline__ v8f vzero8() {
  v8f z;
#pragma unroll
  for (int i = 0; i < 8; ++i) z[i] = 0.0f;
  return z;
}

// D = A*B + C, f16 inputs, f32 accumulate (emits v_wmma_f32_16x16x32_f16)
__device__ __forceinline__ v8f wmma_f32_16x16x32(v16h a, v16h b, v8f c) {
  return __builtin_amdgcn_wmma_f32_16x16x32_f16(false, a, false, b, (short)0, c,
                                                false, false);
}

// 16-bit A/B fragment k-mapping: element e -> k = (e&7) + 8*hi + 16*(e>>3)
// Load fragment row (32 halves, contiguous) with two 16B vector loads.
// Works for both global and LDS-resident rows (addrspace carried through).
__device__ __forceinline__ v16h frag16_from_f16(const _Float16* base, int hi) {
  v8h lo = *(const v8h*)(base + hi * 8);
  v8h up = *(const v8h*)(base + 16 + hi * 8);
  v16h r;
#pragma unroll
  for (int i = 0; i < 8; ++i) { r[i] = lo[i]; r[i + 8] = up[i]; }
  return r;
}

// Same mapping from a contiguous f32 row (on-the-fly f32->f16)
__device__ __forceinline__ v16h frag16_from_f32(const float* base, int hi) {
  v16h r;
#pragma unroll
  for (int e = 0; e < 8; ++e)  r[e]     = (_Float16)base[hi * 8 + e];
#pragma unroll
  for (int e = 0; e < 8; ++e)  r[e + 8] = (_Float16)base[16 + hi * 8 + e];
  return r;
}

// B-fragment gather from a row-major [K x Nout] f32 weight matrix (used only
// by the one-time pre-swizzle kernel).
__device__ __forceinline__ v16h fragB_weight(const float* W, int ldn, int k0,
                                             int ncol, int hi) {
  const float* p = W + (size_t)k0 * ldn + ncol;
  v16h r;
#pragma unroll
  for (int e = 0; e < 16; ++e) {
    int k = (e & 7) + hi * 8 + ((e >> 3) << 4);
    r[e] = (_Float16)p[(size_t)k * ldn];
  }
  return r;
}

// Load a pre-swizzled B-fragment: 32B contiguous per lane (two b128 loads).
__device__ __forceinline__ v16h frag16_swz(const _Float16* wswz, int tile,
                                           int ks, int lane) {
  const _Float16* p = wswz + (((size_t)tile * 8 + ks) * 32 + lane) * 16;
  v8h lo = *(const v8h*)p;
  v8h up = *(const v8h*)(p + 8);
  v16h r;
#pragma unroll
  for (int i = 0; i < 8; ++i) { r[i] = lo[i]; r[i + 8] = up[i]; }
  return r;
}

// Stage one 64B row (32 halves) into LDS with 4 per-lane async b128 copies.
// IOFFSET is added to BOTH the LDS address and the global address, and the
// 16B chunk stride is identical on both sides, so one (vdst, voffset) pair
// covers the whole row.  Tracked with ASYNCcnt (4 per call).
__device__ __forceinline__ void async_stage_row64(uint32_t lds_byte,
                                                  const _Float16* gbase,
                                                  uint32_t voff) {
  asm volatile(
      "global_load_async_to_lds_b128 %0, %1, %2\n\t"
      "global_load_async_to_lds_b128 %0, %1, %2 offset:16\n\t"
      "global_load_async_to_lds_b128 %0, %1, %2 offset:32\n\t"
      "global_load_async_to_lds_b128 %0, %1, %2 offset:48"
      :: "v"(lds_byte), "v"(voff), "s"(gbase)
      : "memory");
}

// ---------------------------------------------------------------------------
// Kernel 0: one-time weight pre-swizzle.  Converts Wf/Wg/Wh (f32, row-major
// [K x Nout]) into f16 B-fragment order: fragment (tile, ks) stores each
// lane's 16 halves contiguously -> kernel 1 loads a fragment with 2 b128s.
// Tiles: 0-1 = Wf, 2-3 = Wg, 4-19 = Wh.  One block per tile, wave = ks.
// ---------------------------------------------------------------------------
__global__ __launch_bounds__(256, 1) void sagan_wswz_kernel(
    const float* __restrict__ Wf, const float* __restrict__ Wg,
    const float* __restrict__ Wh, _Float16* __restrict__ wswz) {
  const int tile = blockIdx.x;          // 0..19
  const int ks   = threadIdx.x >> 5;    // 0..7 (k-step)
  const int lane = threadIdx.x & 31;
  const int hi   = lane >> 4;
  const int lm   = lane & 15;

  const float* W; int ldn, n0;
  if (tile < 2)      { W = Wf; ldn = CKP; n0 = tile * 16; }
  else if (tile < 4) { W = Wg; ldn = CKP; n0 = (tile - 2) * 16; }
  else               { W = Wh; ldn = CIN; n0 = (tile - 4) * 16; }

  v16h r = fragB_weight(W, ldn, ks * 32, n0 + lm, hi);
  _Float16* p = wswz + (((size_t)tile * 8 + ks) * 32 + lane) * 16;
  v8h lo, up;
#pragma unroll
  for (int i = 0; i < 8; ++i) { lo[i] = r[i]; up[i] = r[i + 8]; }
  *(v8h*)p = lo;
  *(v8h*)(p + 8) = up;
}

// ---------------------------------------------------------------------------
// Kernel 1: projections.  f,g stored row-major f16 [TOTTOK,32];
// h stored transposed f16 hT [256, TOTTOK] so the attention loop gets
// contiguous vector loads for its A-fragments.  Weight fragments come from
// the pre-swizzled buffer (2 b128 loads each).
// 256 threads = 8 waves, each wave computes 16 token rows.
// ---------------------------------------------------------------------------
__global__ __launch_bounds__(256, 1) void sagan_proj_kernel(
    const float* __restrict__ x,
    const _Float16* __restrict__ wswz,
    const float* __restrict__ bf, const float* __restrict__ bg,
    const float* __restrict__ bh,
    _Float16* __restrict__ fbuf, _Float16* __restrict__ gbuf,
    _Float16* __restrict__ hT) {
  const int lane = threadIdx.x & 31;
  const int wave = threadIdx.x >> 5;
  const int hi   = lane >> 4;
  const int lm   = lane & 15;
  const int r0   = (blockIdx.x * 8 + wave) * 16;   // token row base

  // Keep all 8 K-step A-fragments of x resident (64 VGPRs), reused by 20 tiles
  v16h xa[8];
  const float* xrow = x + (size_t)(r0 + lm) * CIN;
#pragma unroll
  for (int ks = 0; ks < 8; ++ks) xa[ks] = frag16_from_f32(xrow + ks * 32, hi);

  // f and g projections: 2 column tiles each (swizzle tiles 0-1 / 2-3)
#pragma unroll
  for (int t = 0; t < 2; ++t) {
    v8f af = vzero8(), ag = vzero8();
#pragma unroll
    for (int ks = 0; ks < 8; ++ks) {
      af = wmma_f32_16x16x32(xa[ks], frag16_swz(wswz, t, ks, lane), af);
      ag = wmma_f32_16x16x32(xa[ks], frag16_swz(wswz, 2 + t, ks, lane), ag);
    }
    const float biasf = bf[t * 16 + lm];
    const float biasg = bg[t * 16 + lm];
#pragma unroll
    for (int v = 0; v < 8; ++v) {
      const int row = r0 + v + hi * 8;
      fbuf[(size_t)row * CKP + t * 16 + lm] = (_Float16)(af[v] + biasf);
      gbuf[(size_t)row * CKP + t * 16 + lm] = (_Float16)(ag[v] + biasg);
    }
  }

  // h projection: 16 column tiles (swizzle tiles 4..19), transposed store
  for (int t = 0; t < 16; ++t) {
    v8f ah = vzero8();
#pragma unroll
    for (int ks = 0; ks < 8; ++ks)
      ah = wmma_f32_16x16x32(xa[ks], frag16_swz(wswz, 4 + t, ks, lane), ah);
    const float biash = bh[t * 16 + lm];
    v8h pk;
#pragma unroll
    for (int v = 0; v < 8; ++v) pk[v] = (_Float16)(ah[v] + biash);
    // C/D tile rows v+8*hi are 8 consecutive tokens -> one packed 16B store
    *(v8h*)(hT + (size_t)(t * 16 + lm) * TOTTOK + r0 + hi * 8) = pk;
  }
}

// ---------------------------------------------------------------------------
// Kernel 2: fused flash-attention with async-LDS double buffering of the
// shared h^T key-block tile (16KB per block, staged once for all 8 waves,
// rows padded to 80B for conflict-free b128 LDS reads).
// s^T tiles (keys x queries) put exp(s) directly in B-fragment layout for
// out^T = h^T * p^T (zero cross-lane movement).  Each wave: 16 q x 128 ch.
// ---------------------------------------------------------------------------
__global__ __launch_bounds__(256, 1) void sagan_attn_kernel(
    const float* __restrict__ x, const _Float16* __restrict__ fbuf,
    const _Float16* __restrict__ gbuf, const _Float16* __restrict__ hT,
    const float* __restrict__ gamma, float* __restrict__ out) {
  const int lane = threadIdx.x & 31;
  const int wave = threadIdx.x >> 5;
  const int hi   = lane >> 4;
  const int lm   = lane & 15;
  const int b    = blockIdx.x >> 6;              // 64 blocks per batch
  const int qblk = blockIdx.x & 63;
  const int q0   = qblk * 64 + (wave >> 1) * 16; // query base (in batch)
  const int c0   = (wave & 1) * 128;             // channel half base
  const size_t tokBase = (size_t)b * NTOK;

  // Double-buffered h^T key-block tile: [buf][channel 0..255][key + pad]
  __shared__ __align__(16) _Float16 hsh[2][256][LDSPITCH];

  // Each of the 256 threads owns one 64B channel row of the staged tile.
  const int row = (wave << 5) + lane;                       // 0..255
  const uint32_t rowPart =
      (uint32_t)row * (uint32_t)(TOTTOK * 2) + (uint32_t)(tokBase * 2);
  uint32_t ldsBase[2];
  ldsBase[0] = (uint32_t)(uintptr_t)&hsh[0][row][0];
  ldsBase[1] = (uint32_t)(uintptr_t)&hsh[1][row][0];

  // g B-fragment: [ch x query], loaded once (K = CK = 32, a single fragment)
  const v16h gB = frag16_from_f16(gbuf + (tokBase + q0 + lm) * CKP, hi);

  v8f acc[8];
#pragma unroll
  for (int t = 0; t < 8; ++t) acc[t] = vzero8();

  float m_run = -3.0e38f;
  float l_run = 0.0f;

  // Prologue: stage key block 0 into buffer 0 (ASYNCcnt = 4)
  async_stage_row64(ldsBase[0], hT, rowPart);

  for (int s = 0; s < NTOK / 32; ++s) {
    const int j0 = s << 5;
    // Issue next block (wraps at the end so the wait bound stays uniform and
    // the final block is guaranteed complete; wrapped data is never read).
    const int jn = ((s + 1) & (NTOK / 32 - 1)) << 5;
    async_stage_row64(ldsBase[(s + 1) & 1], hT, rowPart + (uint32_t)(jn * 2));

    // --- score phase (global f only; overlaps the async copies) ---
    const v16h fA0 = frag16_from_f16(fbuf + (tokBase + j0 + lm) * CKP, hi);
    const v16h fA1 = frag16_from_f16(fbuf + (tokBase + j0 + 16 + lm) * CKP, hi);
    v8f s0 = wmma_f32_16x16x32(fA0, gB, vzero8());
    v8f s1 = wmma_f32_16x16x32(fA1, gB, vzero8());

    // Online softmax over the key axis: per-lane reduce + one half-wave merge
    float mx = s0[0];
#pragma unroll
    for (int v = 1; v < 8; ++v) mx = fmaxf(mx, s0[v]);
#pragma unroll
    for (int v = 0; v < 8; ++v) mx = fmaxf(mx, s1[v]);
    mx = fmaxf(mx, __shfl_xor(mx, 16, 32));
    const float m_new = fmaxf(m_run, mx);
    const float scale = __expf(m_run - m_new);

    // exp(s - m): C-layout of s^T tiles == B-fragment layout, pure in-lane
    v16h pB;
    float rsum = 0.0f;
#pragma unroll
    for (int v = 0; v < 8; ++v) {
      const float p0 = __expf(s0[v] - m_new);   // key j0 + v + 8*hi
      const float p1 = __expf(s1[v] - m_new);   // key j0 + 16 + v + 8*hi
      rsum += p0 + p1;
      pB[v]     = (_Float16)p0;
      pB[v + 8] = (_Float16)p1;
    }
    rsum += __shfl_xor(rsum, 16, 32);
    l_run = l_run * scale + rsum;
    m_run = m_new;

    // --- wait for this block's staged h^T tile, then accumulate ---
    asm volatile("s_wait_asynccnt 0x4" ::: "memory");
    __syncthreads();

    const _Float16* hbuf = &hsh[s & 1][0][0];
#pragma unroll
    for (int t = 0; t < 8; ++t) {
      v8f a = acc[t];
#pragma unroll
      for (int v = 0; v < 8; ++v) a[v] *= scale;
      const v16h hA =
          frag16_from_f16(hbuf + (c0 + t * 16 + lm) * LDSPITCH, hi);
      acc[t] = wmma_f32_16x16x32(hA, pB, a);
    }
    __syncthreads();   // all reads done before this buffer is overwritten
  }

  // Epilogue: out = gamma * acc / l + x  (8 contiguous floats per lane/tile)
  const float gm    = gamma[0];
  const float inv_l = 1.0f / l_run;
  const size_t tok  = tokBase + q0 + lm;
#pragma unroll
  for (int t = 0; t < 8; ++t) {
    const int ch = c0 + t * 16 + hi * 8;
    const float* xr = x + tok * CIN + ch;
    float* orw = out + tok * CIN + ch;
#pragma unroll
    for (int v = 0; v < 8; ++v)
      orw[v] = gm * acc[t][v] * inv_l + xr[v];
  }
}

extern "C" void kernel_launch(void* const* d_in, const int* in_sizes, int n_in,
                              void* d_out, int out_size, void* d_ws, size_t ws_size,
                              hipStream_t stream) {
  const float* x     = (const float*)d_in[0];
  const float* Wf    = (const float*)d_in[1];
  const float* bf    = (const float*)d_in[2];
  const float* Wg    = (const float*)d_in[3];
  const float* bg    = (const float*)d_in[4];
  const float* Wh    = (const float*)d_in[5];
  const float* bh    = (const float*)d_in[6];
  const float* gamma = (const float*)d_in[7];
  float* out = (float*)d_out;

  // Workspace (f16): f [32768,32] | g [32768,32] | hT [256,32768] | wswz 160KB
  _Float16* fbuf = (_Float16*)d_ws;
  _Float16* gbuf = fbuf + (size_t)TOTTOK * CKP;
  _Float16* hT   = gbuf + (size_t)TOTTOK * CKP;
  _Float16* wswz = hT + (size_t)CIN * TOTTOK;

  // One-time weight pre-swizzle: 20 tiles x 8 k-steps (f16 B-fragment order)
  sagan_wswz_kernel<<<20, 256, 0, stream>>>(Wf, Wg, Wh, wswz);
  // 256 blocks x 8 waves x 16 rows = 32768 token rows
  sagan_proj_kernel<<<256, 256, 0, stream>>>(x, wswz, bf, bg, bh,
                                             fbuf, gbuf, hT);
  // 8 batches x 64 query-blocks; each block: 8 waves (4 query groups x 2 ch halves)
  sagan_attn_kernel<<<512, 256, 0, stream>>>(x, fbuf, gbuf, hT, gamma, out);
}
